// AFT_FULL_51926154608778
// MI455X (gfx1250) — matrix operand order repaired and checked by reading, not compiled, not run on hardware
//
#include <hip/hip_runtime.h>
#include <hip/hip_bf16.h>
#include <cmath>

#define BS  2
#define SEQ 512
#define DIM 128

typedef __attribute__((ext_vector_type(16))) _Float16 v16h;
typedef __attribute__((ext_vector_type(8)))  _Float16 v8h;
typedef __attribute__((ext_vector_type(8)))  float    v8f;

// ---- WMMA fragment helpers (wave32, V_WMMA_F32_16X16X32_F16) -------------
//
// A (16x32 f16, M x K), row-major memory with K contiguous:
//   lane L = 16*hs + r holds row M=r;
//   v16h elems 0..7  = K[ 16*hs .. 16*hs+7 ]    (16B contiguous)
//   v16h elems 8..15 = K[ 16*hs+8 .. 16*hs+15 ] (16B contiguous)
__device__ __forceinline__ v16h load_frag_a(const _Float16* rowk0, int hs) {
  v8h lo = *reinterpret_cast<const v8h*>(rowk0 + 16 * hs);
  v8h hi = *reinterpret_cast<const v8h*>(rowk0 + 16 * hs + 8);
  v16h r;
#pragma unroll
  for (int i = 0; i < 8; ++i) { r[i] = lo[i]; r[i + 8] = hi[i]; }
  return r;
}

// B (32x16 f16, K x N) from K-contiguous (transposed) storage:
//   lane L = 16*hs + n holds column N=n, K[16*hs .. 16*hs+15] -> one 32B load.
//   Caller passes &T[col*stride + k0 + 16*hs].
__device__ __forceinline__ v16h load_frag_b(const _Float16* p) {
  return *reinterpret_cast<const v16h*>(p);
}

__device__ __forceinline__ v8f wmma_f16(v16h a, v16h b, v8f c) {
  // 8 args: (neg_a, A, neg_b, B, c_mod, C, reuse_a, reuse_b)
  return __builtin_amdgcn_wmma_f32_16x16x32_f16(false, a, false, b, (short)0, c,
                                                false, false);
}

// ---- prep kernels ---------------------------------------------------------

__global__ void k_cvt_x(const float* __restrict__ x, _Float16* __restrict__ xh, int n) {
  int i = blockIdx.x * blockDim.x + threadIdx.x;
  if (i < n) xh[i] = (_Float16)x[i];
}

// Transpose + convert all three weight matrices: wT[n*DIM+k] = W[k*DIM+n]
__global__ void k_cvt_wT(const float* __restrict__ wq, const float* __restrict__ wk,
                         const float* __restrict__ wv, _Float16* __restrict__ wqT,
                         _Float16* __restrict__ wkT, _Float16* __restrict__ wvT, int n) {
  int i = blockIdx.x * blockDim.x + threadIdx.x;
  if (i >= n) return;
  int sel = i / (DIM * DIM);
  int rem = i % (DIM * DIM);
  int nn = rem / DIM, kk = rem % DIM;
  const float* w = (sel == 0) ? wq : (sel == 1) ? wk : wv;
  _Float16*   wt = (sel == 0) ? wqT : (sel == 1) ? wkT : wvT;
  wt[nn * DIM + kk] = (_Float16)w[kk * DIM + nn];
}

__global__ void k_exp_b(const float* __restrict__ B, _Float16* __restrict__ ebh, int n) {
  int i = blockIdx.x * blockDim.x + threadIdx.x;
  if (i < n) ebh[i] = (_Float16)expf(B[i]);
}

// ---- kernel 1: fused QKV GEMM + epilogue ---------------------------------
// One wave per 16x16 output tile of the 1024x128 projection.
// Reuses the A (x) fragments across Wq/Wk/Wv. Epilogue writes
//   sq  = sigmoid(q)              [b][j][d]  f32
//   ekT = exp(k)                  [b][d][j]  f16  (B-operand layout for stage 2)
//   pT  = exp(k)*v                [b][d][j]  f16
__global__ void __launch_bounds__(256) k_qkv(
    const _Float16* __restrict__ xh, const _Float16* __restrict__ wqT,
    const _Float16* __restrict__ wkT, const _Float16* __restrict__ wvT,
    const float* __restrict__ bq, const float* __restrict__ bk,
    const float* __restrict__ bv, float* __restrict__ sq,
    _Float16* __restrict__ ekT, _Float16* __restrict__ pT) {
  const int lane = threadIdx.x & 31;
  const int wid  = (blockIdx.x * blockDim.x + threadIdx.x) >> 5;  // 0..511
  const int mt = wid >> 3, nt = wid & 7;
  const int m0 = mt * 16, n0 = nt * 16;
  const int r = lane & 15, hs = lane >> 4;

  // A fragments for all 4 K-steps (K = 128)
  v16h a[4];
#pragma unroll
  for (int kk = 0; kk < 4; ++kk)
    a[kk] = load_frag_a(xh + (size_t)(m0 + r) * DIM + kk * 32, hs);

  const int col = n0 + r;         // output column (d) owned by this lane
  const int bko = 16 * hs;        // B-fragment K sub-offset

  v8f cq = {}, ck = {}, cv = {};
#pragma unroll
  for (int kk = 0; kk < 4; ++kk)
    cq = wmma_f16(a[kk], load_frag_b(wqT + (size_t)col * DIM + kk * 32 + bko), cq);
#pragma unroll
  for (int kk = 0; kk < 4; ++kk)
    ck = wmma_f16(a[kk], load_frag_b(wkT + (size_t)col * DIM + kk * 32 + bko), ck);
#pragma unroll
  for (int kk = 0; kk < 4; ++kk)
    cv = wmma_f16(a[kk], load_frag_b(wvT + (size_t)col * DIM + kk * 32 + bko), cv);

  const float biasq = bq[col], biask = bk[col], biasv = bv[col];
#pragma unroll
  for (int g = 0; g < 8; ++g) {
    int gm = m0 + g + 8 * hs;          // flattened (b, j) row: 0..1023
    int b  = gm >> 9;
    int j  = gm & (SEQ - 1);
    float qv = cq[g] + biasq;
    sq[((size_t)b * SEQ + j) * DIM + col] = 1.0f / (1.0f + expf(-qv));
    float ek = expf(ck[g] + biask);
    float pv = ek * (cv[g] + biasv);
    size_t ti = ((size_t)b * DIM + col) * SEQ + j;   // transposed [b][d][j]
    ekT[ti] = (_Float16)ek;
    pT[ti]  = (_Float16)pv;
  }
}

// ---- kernel 2: AFT GEMMs  num = EB @ P, den = EB @ EK --------------------
// One wave per (batch, 16-row query tile, 16-col d tile). K = 512.
// A (EB tile) fragment reused for both num and den accumulators.
__global__ void __launch_bounds__(256) k_aft(
    const _Float16* __restrict__ ebh, const _Float16* __restrict__ ekT,
    const _Float16* __restrict__ pT, const float* __restrict__ sq,
    float* __restrict__ out) {
  const int lane = threadIdx.x & 31;
  const int wid  = (blockIdx.x * blockDim.x + threadIdx.x) >> 5;  // 0..511
  const int b  = wid >> 8;
  const int it = (wid >> 3) & 31;
  const int dt = wid & 7;
  const int i0 = it * 16, d0 = dt * 16;
  const int r = lane & 15, hs = lane >> 4;

  const _Float16* arow = ebh + (size_t)(i0 + r) * SEQ;
  const _Float16* bnum = pT  + ((size_t)b * DIM + d0 + r) * SEQ + 16 * hs;
  const _Float16* bden = ekT + ((size_t)b * DIM + d0 + r) * SEQ + 16 * hs;

  v8f cn = {}, cd = {};
#pragma unroll 4
  for (int j0 = 0; j0 < SEQ; j0 += 32) {
    __builtin_prefetch(arow + j0 + 128, 0, 1);     // global_prefetch_b8
    v16h a  = load_frag_a(arow + j0, hs);
    v16h fn = load_frag_b(bnum + j0);
    v16h fd = load_frag_b(bden + j0);
    cn = wmma_f16(a, fn, cn);
    cd = wmma_f16(a, fd, cd);
  }

#pragma unroll
  for (int g = 0; g < 8; ++g) {
    int i = i0 + g + 8 * hs;
    int d = d0 + r;
    size_t idx = ((size_t)b * SEQ + i) * DIM + d;
    out[idx] = sq[idx] * (cn[g] / cd[g]);
  }
}

// ---- host launch ----------------------------------------------------------

extern "C" void kernel_launch(void* const* d_in, const int* in_sizes, int n_in,
                              void* d_out, int out_size, void* d_ws, size_t ws_size,
                              hipStream_t stream) {
  (void)in_sizes; (void)n_in; (void)out_size; (void)ws_size;
  const float* x  = (const float*)d_in[0];
  const float* Wq = (const float*)d_in[1];
  const float* bq = (const float*)d_in[2];
  const float* Wk = (const float*)d_in[3];
  const float* bk = (const float*)d_in[4];
  const float* Wv = (const float*)d_in[5];
  const float* bv = (const float*)d_in[6];
  const float* B  = (const float*)d_in[7];
  float* out = (float*)d_out;

  // Carve workspace (~1.9 MB total; everything stays L2-resident).
  char* ws = (char*)d_ws;
  size_t off = 0;
  auto carve = [&](size_t bytes) -> char* {
    char* p = ws + off;
    off += (bytes + 255) & ~(size_t)255;
    return p;
  };
  float*    sq  = (float*)   carve((size_t)BS * SEQ * DIM * sizeof(float));
  _Float16* xh  = (_Float16*)carve((size_t)BS * SEQ * DIM * sizeof(_Float16));
  _Float16* wqT = (_Float16*)carve((size_t)DIM * DIM * sizeof(_Float16));
  _Float16* wkT = (_Float16*)carve((size_t)DIM * DIM * sizeof(_Float16));
  _Float16* wvT = (_Float16*)carve((size_t)DIM * DIM * sizeof(_Float16));
  _Float16* ebh = (_Float16*)carve((size_t)SEQ * SEQ * sizeof(_Float16));
  _Float16* ekT = (_Float16*)carve((size_t)BS * DIM * SEQ * sizeof(_Float16));
  _Float16* pT  = (_Float16*)carve((size_t)BS * DIM * SEQ * sizeof(_Float16));

  const int nx  = BS * SEQ * DIM;   // 131072
  const int nw  = 3 * DIM * DIM;    // 49152
  const int nb  = SEQ * SEQ;        // 262144

  k_cvt_x <<<(nx + 255) / 256, 256, 0, stream>>>(x, xh, nx);
  k_cvt_wT<<<(nw + 255) / 256, 256, 0, stream>>>(Wq, Wk, Wv, wqT, wkT, wvT, nw);
  k_exp_b <<<(nb + 255) / 256, 256, 0, stream>>>(B, ebh, nb);

  // 512 waves each (one per 16x16 tile), 8 waves / 256-thread block.
  k_qkv<<<64, 256, 0, stream>>>(xh, wqT, wkT, wvT, bq, bk, bv, sq, ekT, pT);
  k_aft<<<64, 256, 0, stream>>>(ebh, ekT, pT, sq, out);
}
